// Multi_head_attention_88768384074134
// MI455X (gfx1250) — matrix operand order repaired and checked by reading, not compiled, not run on hardware
//
#include <hip/hip_runtime.h>
#include <hip/hip_bf16.h>

// ---------------------------------------------------------------------------
// Multi-head attention with shared head projections, CDNA5 (gfx1250) WMMA.
//   B=4, S=4096, D=512, H=8, DK=64.  All f16 WMMA with f32 accumulation.
// ---------------------------------------------------------------------------

typedef __attribute__((ext_vector_type(16))) _Float16 v16h;
typedef __attribute__((ext_vector_type(8)))  float    v8f;
typedef __attribute__((ext_vector_type(4)))  int      i32x4;
typedef _Float16 f16;

#define BB 4
#define SS 4096
#define DD 512
#define DKK 64
#define NR (BB*SS)            // 16384 rows total
#define ROWCH (NR*DKK)        // 1048576 halves per [NR x 64] buffer

// Optional CDNA5 async global->LDS copy (ASYNCcnt-tracked); guarded so the
// build cannot regress on toolchains that do not declare the builtins.
#if defined(__has_builtin)
#if __has_builtin(__builtin_amdgcn_global_load_async_to_lds_b128) && \
    __has_builtin(__builtin_amdgcn_s_wait_asynccnt)
#define HAVE_ASYNC_LDS 1
#endif
#endif
#ifndef HAVE_ASYNC_LDS
#define HAVE_ASYNC_LDS 0
#endif

static __device__ __forceinline__ v8f wmma_f16(v16h a, v16h b, v8f c) {
    return __builtin_amdgcn_wmma_f32_16x16x32_f16(
        /*neg_a=*/false, a, /*neg_b=*/false, b,
        /*c_mod=*/(short)0, c, /*reuse_a=*/false, /*reuse_b=*/false);
}

// copy 16 bytes global -> LDS
static __device__ __forceinline__ void copy16_g2l(const f16* g, f16* l) {
#if HAVE_ASYNC_LDS
    __builtin_amdgcn_global_load_async_to_lds_b128(
        (__attribute__((address_space(1))) i32x4*)g,
        (__attribute__((address_space(3))) i32x4*)l, 0, 0);
#else
    typedef __attribute__((ext_vector_type(8))) _Float16 v8h;
    *(v8h*)l = *(const v8h*)g;
#endif
}

static __device__ __forceinline__ void async_drain() {
#if HAVE_ASYNC_LDS
    __builtin_amdgcn_s_wait_asynccnt(0);
#endif
}

// ---------------------------------------------------------------------------
// Kernel 0: weights -> f16, transposed so WMMA B-fragments are contiguous.
//   Wqt/Wkt/Wvt : [64][512]   Wqt[n*512+k] = Wq[k*64+n]
//   Woet        : [512][64]   Woet[n*64+k] = sum_h Wo[(h*64+k)*512+n]
// ---------------------------------------------------------------------------
__global__ void __launch_bounds__(256)
prep_weights(const float* __restrict__ Wq, const float* __restrict__ Wk,
             const float* __restrict__ Wv, const float* __restrict__ Wo,
             f16* __restrict__ Wqt, f16* __restrict__ Wkt,
             f16* __restrict__ Wvt, f16* __restrict__ Woet) {
    int id = blockIdx.x * blockDim.x + threadIdx.x;   // 0 .. 131071
    int sel = id >> 15;                               // which matrix
    int idx = id & 32767;
    if (sel < 3) {
        int n = idx >> 9;          // 0..63
        int k = idx & 511;         // 0..511
        const float* src = (sel == 0) ? Wq : (sel == 1) ? Wk : Wv;
        f16* dst = (sel == 0) ? Wqt : (sel == 1) ? Wkt : Wvt;
        dst[idx] = (f16)src[k * DKK + n];
    } else {
        int n = idx >> 6;          // 0..511
        int k = idx & 63;          // 0..63
        float s = 0.f;
        #pragma unroll
        for (int h = 0; h < 8; ++h) s += Wo[(size_t)(h * 64 + k) * DD + n];
        Woet[idx] = (f16)s;
    }
}

// ---------------------------------------------------------------------------
// Kernel 1: QKV projection.  [16384x512]x[512x64] for q,k,v simultaneously.
// One wave per 16-row tile; 8 waves (256 threads) per block -> 128 rows/block.
// ---------------------------------------------------------------------------
__global__ void __launch_bounds__(256)
qkv_proj(const float* __restrict__ x,
         const f16* __restrict__ Wqt, const f16* __restrict__ Wkt,
         const f16* __restrict__ Wvt,
         const float* __restrict__ bq, const float* __restrict__ bk,
         const float* __restrict__ bv,
         f16* __restrict__ Qh, f16* __restrict__ Kh, f16* __restrict__ Vh) {
    const int lane = threadIdx.x & 31;
    const int wave = threadIdx.x >> 5;
    const int g    = lane >> 4;       // half-wave group
    const int ln   = lane & 15;
    const int row0 = blockIdx.x * 128 + wave * 16;

    v8f acc[3][4] = {};               // [proj][n-tile], n-tiles cover DK=64

    const float* xp = x + (size_t)(row0 + ln) * DD;
    const f16* Wt[3] = {Wqt, Wkt, Wvt};

    for (int kk = 0; kk < DD; kk += 32) {
        // A fragment: rows of x, f32 -> f16 in-register (k = 16g + s, matches B)
        v16h a;
        #pragma unroll
        for (int i = 0; i < 8; ++i) {
            a[i]     = (f16)xp[kk + g * 16 + i];
            a[i + 8] = (f16)xp[kk + g * 16 + 8 + i];
        }
        #pragma unroll
        for (int p = 0; p < 3; ++p) {
            #pragma unroll
            for (int nt = 0; nt < 4; ++nt) {
                const f16* wp = Wt[p] + (size_t)(nt * 16 + ln) * DD + kk + g * 16;
                v16h bfr = *(const v16h*)wp;              // contiguous 32B
                acc[p][nt] = wmma_f16(a, bfr, acc[p][nt]);
            }
        }
    }

    // epilogue: + bias, f32 -> f16, store (C layout: M = j + 8g, N = lane%16)
    const float* bias[3] = {bq, bk, bv};
    f16* out[3] = {Qh, Kh, Vh};
    #pragma unroll
    for (int p = 0; p < 3; ++p) {
        #pragma unroll
        for (int nt = 0; nt < 4; ++nt) {
            int n = nt * 16 + ln;
            float bb = bias[p][n];
            #pragma unroll
            for (int j = 0; j < 8; ++j) {
                int r = row0 + j + 8 * g;
                out[p][(size_t)r * DKK + n] = (f16)(acc[p][nt][j] + bb);
            }
        }
    }
}

// ---------------------------------------------------------------------------
// Kernel 2: flash attention.  grid = (S/128, B); 8 waves/block, each wave owns
// a 16-query tile; key blocks of 64 are staged once per block in LDS (K
// verbatim, V transposed) so all WMMA B-fragments are contiguous LDS reads.
// ---------------------------------------------------------------------------
__global__ void __launch_bounds__(256)
flash_attn(const f16* __restrict__ Qh, const f16* __restrict__ Kh,
           const f16* __restrict__ Vh, f16* __restrict__ Oh) {
    __shared__ __align__(32) f16 Klds[64 * 64];       // [key][dk]   8 KB
    __shared__ __align__(32) f16 Vt[64 * 64];         // [dk][key]   8 KB
    __shared__ __align__(32) f16 ldsP[8][16 * 64];    // per-wave P  16 KB

    const int tid  = threadIdx.x;
    const int lane = tid & 31;
    const int wave = tid >> 5;
    const int g    = lane >> 4;
    const int ln   = lane & 15;
    const int b    = blockIdx.y;
    const int q0   = blockIdx.x * 128 + wave * 16;     // query row in batch
    const size_t base = (size_t)b * SS * DKK;

    // staging assignment: thread copies 32B of K and 32B of V per key block
    const int srow = tid >> 2;             // 0..63  (key row)
    const int sseg = (tid & 3) * 16;       // 0/16/32/48 (dk start, halves)

    // Q A-fragments for k-steps 0 and 32 of DK (loaded once)
    v16h aq[2];
    {
        const f16* qp = Qh + base + (size_t)(q0 + ln) * DKK;
        #pragma unroll
        for (int kk2 = 0; kk2 < 2; ++kk2)
            #pragma unroll
            for (int i = 0; i < 8; ++i) {
                aq[kk2][i]     = qp[kk2 * 32 + g * 16 + i];
                aq[kk2][i + 8] = qp[kk2 * 32 + g * 16 + 8 + i];
            }
    }

    v8f o[4] = {};
    float mx[8], lsum[8];
    #pragma unroll
    for (int j = 0; j < 8; ++j) { mx[j] = -1e30f; lsum[j] = 0.f; }

    for (int kb = 0; kb < SS; kb += 64) {
        // ---- stage K tile (verbatim) and V tile (transposed) into LDS ----
        {
            const f16* ks = Kh + base + (size_t)(kb + srow) * DKK + sseg;
            copy16_g2l(ks,     &Klds[srow * 64 + sseg]);
            copy16_g2l(ks + 8, &Klds[srow * 64 + sseg + 8]);

            const f16* vs = Vh + base + (size_t)(kb + srow) * DKK + sseg;
            v16h vv = *(const v16h*)vs;                  // 32B global read
            #pragma unroll
            for (int i = 0; i < 16; ++i)                 // scatter-transpose
                Vt[(sseg + i) * 64 + srow] = vv[i];
            async_drain();
        }
        __syncthreads();

        if (kb + 64 < SS) {   // hint next K block toward the caches
            __builtin_prefetch((const void*)(Kh + base + (size_t)(kb + 64 + srow) * DKK), 0, 1);
        }

        // ---- scores S = Q K^T (B-fragments straight from Klds) ----
        v8f s[4] = {};
        #pragma unroll
        for (int nt = 0; nt < 4; ++nt) {
            const f16* kp = &Klds[(nt * 16 + ln) * 64 + g * 16];
            v16h b0 = *(const v16h*)kp;            // dk 0..31 slice
            v16h b1 = *(const v16h*)(kp + 32);     // dk 32..63 slice
            s[nt] = wmma_f16(aq[0], b0, s[nt]);
            s[nt] = wmma_f16(aq[1], b1, s[nt]);
        }
        #pragma unroll
        for (int nt = 0; nt < 4; ++nt)
            #pragma unroll
            for (int j = 0; j < 8; ++j) s[nt][j] *= 0.125f;   // 1/sqrt(64)

        // ---- online softmax (row = j + 8g; 16 lanes share a row) ----
        #pragma unroll
        for (int j = 0; j < 8; ++j) {
            float bm = fmaxf(fmaxf(s[0][j], s[1][j]), fmaxf(s[2][j], s[3][j]));
            #pragma unroll
            for (int m = 1; m <= 8; m <<= 1) bm = fmaxf(bm, __shfl_xor(bm, m));
            float mnew = fmaxf(mx[j], bm);
            float corr = __expf(mx[j] - mnew);
            mx[j] = mnew;
            float rs = 0.f;
            #pragma unroll
            for (int nt = 0; nt < 4; ++nt) {
                float p = __expf(s[nt][j] - mnew);
                rs += p;
                ldsP[wave][(j + 8 * g) * 64 + nt * 16 + ln] = (f16)p;
            }
            #pragma unroll
            for (int m = 1; m <= 8; m <<= 1) rs += __shfl_xor(rs, m);
            lsum[j] = lsum[j] * corr + rs;
            #pragma unroll
            for (int nt = 0; nt < 4; ++nt) o[nt][j] *= corr;
        }

        // ---- O += P V  (P from per-wave LDS strip in A layout; V from
        //      transposed LDS tile, contiguous 32B per B-fragment) ----
        #pragma unroll
        for (int kk2 = 0; kk2 < 2; ++kk2) {
            v16h ap;
            const f16* pp = &ldsP[wave][ln * 64 + kk2 * 32];
            #pragma unroll
            for (int i = 0; i < 8; ++i) {
                ap[i]     = pp[g * 16 + i];
                ap[i + 8] = pp[g * 16 + 8 + i];
            }
            #pragma unroll
            for (int nt2 = 0; nt2 < 4; ++nt2) {
                const f16* vp = &Vt[(nt2 * 16 + ln) * 64 + kk2 * 32 + g * 16];
                v16h bv = *(const v16h*)vp;
                o[nt2] = wmma_f16(ap, bv, o[nt2]);
            }
        }
        __syncthreads();     // protect K/V tiles before next staging pass
    }

    // normalize and store head output (f16)
    #pragma unroll
    for (int nt2 = 0; nt2 < 4; ++nt2) {
        #pragma unroll
        for (int j = 0; j < 8; ++j) {
            int r = q0 + j + 8 * g;
            Oh[base + (size_t)r * DKK + nt2 * 16 + ln] = (f16)(o[nt2][j] / lsum[j]);
        }
    }
}

// ---------------------------------------------------------------------------
// Kernel 3: out = head @ Wo_eff + bo.   [16384x64]x[64x512], f32 output.
// ---------------------------------------------------------------------------
__global__ void __launch_bounds__(256)
out_proj(const f16* __restrict__ Oh, const f16* __restrict__ Woet,
         const float* __restrict__ bo, float* __restrict__ out) {
    const int lane = threadIdx.x & 31;
    const int wave = threadIdx.x >> 5;
    const int g    = lane >> 4;
    const int ln   = lane & 15;
    const int row0 = blockIdx.x * 128 + wave * 16;

    v16h ao[2];
    {
        const f16* op = Oh + (size_t)(row0 + ln) * DKK;
        #pragma unroll
        for (int kk2 = 0; kk2 < 2; ++kk2)
            #pragma unroll
            for (int i = 0; i < 8; ++i) {
                ao[kk2][i]     = op[kk2 * 32 + g * 16 + i];
                ao[kk2][i + 8] = op[kk2 * 32 + g * 16 + 8 + i];
            }
    }

    for (int nt = 0; nt < 32; ++nt) {
        int n = nt * 16 + ln;
        const f16* wp = Woet + (size_t)n * DKK + g * 16;
        v16h b0 = *(const v16h*)wp;
        v16h b1 = *(const v16h*)(wp + 32);
        v8f acc = {};
        acc = wmma_f16(ao[0], b0, acc);
        acc = wmma_f16(ao[1], b1, acc);
        float bb = bo[n];
        #pragma unroll
        for (int j = 0; j < 8; ++j) {
            int r = row0 + j + 8 * g;
            out[(size_t)r * DD + n] = acc[j] + bb;
        }
    }
}

// ---------------------------------------------------------------------------
extern "C" void kernel_launch(void* const* d_in, const int* in_sizes, int n_in,
                              void* d_out, int out_size, void* d_ws, size_t ws_size,
                              hipStream_t stream) {
    const float* x  = (const float*)d_in[0];
    const float* Wq = (const float*)d_in[1];
    const float* bq = (const float*)d_in[2];
    const float* Wk = (const float*)d_in[3];
    const float* bk = (const float*)d_in[4];
    const float* Wv = (const float*)d_in[5];
    const float* bv = (const float*)d_in[6];
    const float* Wo = (const float*)d_in[7];
    const float* bo = (const float*)d_in[8];
    float* out = (float*)d_out;

    // workspace layout (f16): Qh | Kh | Vh | Oh | Wqt | Wkt | Wvt | Woet
    f16* W   = (f16*)d_ws;
    f16* Qh  = W;
    f16* Kh  = W + (size_t)1 * ROWCH;
    f16* Vh  = W + (size_t)2 * ROWCH;
    f16* Oh  = W + (size_t)3 * ROWCH;
    f16* Wqt = W + (size_t)4 * ROWCH;
    f16* Wkt = Wqt + 32768;
    f16* Wvt = Wkt + 32768;
    f16* Woet = Wvt + 32768;

    prep_weights<<<512, 256, 0, stream>>>(Wq, Wk, Wv, Wo, Wqt, Wkt, Wvt, Woet);
    qkv_proj<<<NR / 128, 256, 0, stream>>>(x, Wqt, Wkt, Wvt, bq, bk, bv, Qh, Kh, Vh);
    flash_attn<<<dim3(SS / 128, BB), 256, 0, stream>>>(Qh, Kh, Vh, Oh);
    out_proj<<<NR / 128, 256, 0, stream>>>(Oh, Woet, bo, out);
}
